// MultiHeadSelfAttention_44375602102965
// MI455X (gfx1250) — matrix operand order repaired and checked by reading, not compiled, not run on hardware
//
#include <hip/hip_runtime.h>
#include <hip/hip_bf16.h>
#include <math.h>

// ---------------------------------------------------------------------------
// MI455X (gfx1250) multi-head self-attention, wave32 + WMMA f16->f32.
// Everything (<=24MB) is L2-resident; compute-bound -> all matmuls via
// v_wmma_f32_16x16x32_f16. Softmax in exp2 domain (log2e folded into the Q
// projection scale), stats in f32, exp work spread over all 256 threads.
// ---------------------------------------------------------------------------

typedef __attribute__((ext_vector_type(16))) _Float16 v16h;
typedef __attribute__((ext_vector_type(8)))  _Float16 v8h;
typedef __attribute__((ext_vector_type(8)))  float    v8f;
typedef __attribute__((ext_vector_type(4)))  float    v4f;

constexpr int kB  = 4;
constexpr int kC  = 256;
constexpr int kS  = 2304;   // 48*48
constexpr int kNH = 8;
constexpr int kHD = 32;
// softmax(q.k * 1/sqrt(HD)) computed as exp2((q*QS).k - m):
// QS = log2(e)/sqrt(32)
constexpr float kQScale = 0.17677669529663687f * 1.4426950408889634f;

constexpr int kTK    = 128;  // keys per attention inner tile
constexpr int kPPAD  = 144;  // pbuf row stride in halves (288B, 16B multiple)
constexpr int kScPad = 132;  // sc row stride in floats  (528B, 16B multiple)

// ---- WMMA fragment helpers (layouts per CDNA5 ISA 7.12.2, wave32) ---------
// A 16x32 f16: lane<16 -> row M=lane, K chunks [0..7] and [16..23];
//              lane>=16 -> row M=lane-16, K chunks [8..15] and [24..31].
__device__ __forceinline__ v16h frag_a_f16(const _Float16* p, int ld) {
  const int lane = threadIdx.x & 31;
  const _Float16* base = p + (lane & 15) * ld + ((lane < 16) ? 0 : 8);
  v8h lo = *(const v8h*)(base);
  v8h hi = *(const v8h*)(base + 16);
  v16h r;
#pragma unroll
  for (int i = 0; i < 8; ++i) { r[i] = lo[i]; r[i + 8] = hi[i]; }
  return r;
}

// Same A layout but source is f32 (weights): load 2x8 floats, convert.
__device__ __forceinline__ v16h frag_a_f32(const float* p, int ld) {
  const int lane = threadIdx.x & 31;
  const float* base = p + (lane & 15) * ld + ((lane < 16) ? 0 : 8);
  v16h r;
#pragma unroll
  for (int i = 0; i < 8; ++i) {
    r[i]     = (_Float16)base[i];
    r[i + 8] = (_Float16)base[i + 16];
  }
  return r;
}

// B 32x16 f16: lane<16 -> col N=lane, K chunks [0..7] and [16..23];
//              lane>=16 -> col N=lane-16, K chunks [8..15] and [24..31].
// Memory viewed as rows of length ld indexed by N, K contiguous within a row.
__device__ __forceinline__ v16h frag_b_f16(const _Float16* p, int ld) {
  const int lane = threadIdx.x & 31;
  const _Float16* base = p + (lane & 15) * ld + ((lane < 16) ? 0 : 8);
  v8h lo = *(const v8h*)(base);
  v8h hi = *(const v8h*)(base + 16);
  v16h r;
#pragma unroll
  for (int i = 0; i < 8; ++i) { r[i] = lo[i]; r[i + 8] = hi[i]; }
  return r;
}

__device__ __forceinline__ v8f wmma16(v16h a, v16h b, v8f c) {
  return __builtin_amdgcn_wmma_f32_16x16x32_f16(false, a, false, b, (short)0, c,
                                                false, false);
}

// ---------------------------------------------------------------------------
// Kernel 0: x [B,C,S] f32  ->  xt [B,S,C] f16  (LDS tile transpose)
// ---------------------------------------------------------------------------
__global__ __launch_bounds__(256) void xpose_kernel(const float* __restrict__ x,
                                                    _Float16* __restrict__ xt) {
  __shared__ float t[32][33];
  const int s0 = blockIdx.x * 32;
  const int c0 = blockIdx.y * 32;
  const int b  = blockIdx.z;
#pragma unroll
  for (int i = 0; i < 4; ++i) {
    int c = c0 + threadIdx.y + 8 * i;
    int s = s0 + threadIdx.x;
    t[threadIdx.y + 8 * i][threadIdx.x] = x[((size_t)b * kC + c) * kS + s];
  }
  __syncthreads();
#pragma unroll
  for (int i = 0; i < 4; ++i) {
    int s = s0 + threadIdx.y + 8 * i;
    int c = c0 + threadIdx.x;
    xt[((size_t)b * kS + s) * kC + c] = (_Float16)t[threadIdx.x][threadIdx.y + 8 * i];
  }
}

// ---------------------------------------------------------------------------
// Kernel 1: QKV projection. GEMM: out[o,s] = sum_c W[o,c] * x[b,c,s] + bias.
// blockIdx.z = b*3 + {0:Q,1:K,2:V}. 8 waves per block compute a 64x64 tile.
// Q stored [B,NH,S,HD]*QScale, K stored [B,NH,S,HD], V stored [B,NH,HD,S].
// ---------------------------------------------------------------------------
__global__ __launch_bounds__(256) void qkv_kernel(
    const _Float16* __restrict__ xt,
    const float* __restrict__ Wq, const float* __restrict__ bq,
    const float* __restrict__ Wk, const float* __restrict__ bk,
    const float* __restrict__ Wv, const float* __restrict__ bv,
    _Float16* __restrict__ Qd, _Float16* __restrict__ Kd,
    _Float16* __restrict__ Vtd) {
  const int lane  = threadIdx.x & 31;
  const int w     = threadIdx.x >> 5;
  const int r     = w & 3;          // M (row) tile within the 64x64 block
  const int cpair = w >> 2;         // 0..1, pair of N col tiles
  const int s0 = blockIdx.x * 64;
  const int o0 = blockIdx.y * 64 + 16 * r;
  const int which = blockIdx.z % 3;
  const int b     = blockIdx.z / 3;
  const float* Wm = (which == 0) ? Wq : (which == 1) ? Wk : Wv;
  const float* bm = (which == 0) ? bq : (which == 1) ? bk : bv;
  const _Float16* xb = xt + (size_t)b * kS * kC;

  v8f acc0 = {}, acc1 = {};
  for (int kk = 0; kk < kC; kk += 32) {
    v16h fa  = frag_a_f32(Wm + o0 * kC + kk, kC);
    v16h fb0 = frag_b_f16(xb + (s0 + 32 * cpair) * kC + kk, kC);
    v16h fb1 = frag_b_f16(xb + (s0 + 32 * cpair + 16) * kC + kk, kC);
    acc0 = wmma16(fa, fb0, acc0);
    acc1 = wmma16(fa, fb1, acc1);
  }

  const int hi8  = (lane >> 4) << 3;
  const int ncol = lane & 15;
#pragma unroll
  for (int j = 0; j < 8; ++j) {
    const int o  = o0 + j + hi8;
    const float bias = bm[o];
    const int nh = o >> 5;
    const int d  = o & 31;
#pragma unroll
    for (int ct = 0; ct < 2; ++ct) {
      const int s = s0 + 32 * cpair + 16 * ct + ncol;
      const float v = (ct ? acc1[j] : acc0[j]) + bias;
      if (which == 0)
        Qd[(((size_t)b * kNH + nh) * kS + s) * kHD + d] = (_Float16)(v * kQScale);
      else if (which == 1)
        Kd[(((size_t)b * kNH + nh) * kS + s) * kHD + d] = (_Float16)v;
      else
        Vtd[(((size_t)b * kNH + nh) * kHD + d) * kS + s] = (_Float16)v;
    }
  }
}

// ---------------------------------------------------------------------------
// Kernel 2: flash-style attention. One block = 64 query rows of one (b,head),
// 128-key inner tiles. QK^T: K-dim == HD == 32 -> one WMMA per 16x16 tile.
// Online softmax (exp2 domain): all 256 threads (4 per query row) share the
// exp work; LDS traffic vectorized to b128; running stats (rm, rs) in LDS.
// ---------------------------------------------------------------------------
__global__ __launch_bounds__(256) void attn_kernel(
    const _Float16* __restrict__ Qd, const _Float16* __restrict__ Kd,
    const _Float16* __restrict__ Vtd, _Float16* __restrict__ AO) {
  __shared__ __align__(16) float sc[64][kScPad];        // raw scores (f32)
  __shared__ __align__(16) _Float16 pbuf[64][kPPAD];    // exp probs  (f16)
  __shared__ float pmax[64][4];
  __shared__ float psum[64][4];
  __shared__ float rm_s[64], rs_s[64], corr_s[64];

  const int tid   = threadIdx.x;
  const int lane  = tid & 31;
  const int w     = tid >> 5;
  const int r     = w & 3;       // 16-row tile of the 64 queries
  const int cq    = w >> 2;      // 0..1: 4 score col tiles / 1 PV d tile
  const int s0 = blockIdx.x * 64;
  const int bh = blockIdx.y;     // b*NH + nh
  const _Float16* Qb = Qd + (size_t)bh * kS * kHD;
  const _Float16* Kb = Kd + (size_t)bh * kS * kHD;
  const _Float16* Vb = Vtd + (size_t)bh * kHD * kS;

  const int hi8  = (lane >> 4) << 3;
  const int ncol = lane & 15;
  const int srow = tid >> 2;     // softmax: this thread's query row
  const int sq   = tid & 3;      // softmax: this thread's 32-col quarter

  if (tid < 64) { rm_s[tid] = -1e30f; rs_s[tid] = 0.0f; }

  // Q fragment is loop-invariant: rows s0+16r..+15, all 32 d's.
  const v16h qf = frag_a_f16(Qb + (size_t)(s0 + 16 * r) * kHD, kHD);
  v8f oacc = {};                 // PV accumulator: rows 16r.., cols d=16*cq..

  for (int t0 = 0; t0 < kS; t0 += kTK) {
    // --- scores: this wave's four 16x16 tiles (one WMMA each, K=HD=32) ---
#pragma unroll
    for (int cti = 0; cti < 4; ++cti) {
      const int ct = 4 * cq + cti;
      v16h kf = frag_b_f16(Kb + (size_t)(t0 + 16 * ct) * kHD, kHD);
      v8f acc = {};
      acc = wmma16(qf, kf, acc);
#pragma unroll
      for (int j = 0; j < 8; ++j)
        sc[16 * r + j + hi8][16 * ct + ncol] = acc[j];
    }
    __syncthreads();

    // --- softmax, step 1: partial row max, b128 reads (4 threads/row) ---
    const v4f* scv = (const v4f*)&sc[srow][32 * sq];
    {
      float m = -1e30f;
#pragma unroll
      for (int g = 0; g < 8; ++g) {
        v4f t = scv[g];
        m = fmaxf(m, fmaxf(fmaxf(t[0], t[1]), fmaxf(t[2], t[3])));
      }
      pmax[srow][sq] = m;
    }
    __syncthreads();

    // --- softmax, step 2: exp2 with the new running max, b128 writes ---
    {
      const float nm = fmaxf(rm_s[srow],
                             fmaxf(fmaxf(pmax[srow][0], pmax[srow][1]),
                                   fmaxf(pmax[srow][2], pmax[srow][3])));
      v8h* prow = (v8h*)&pbuf[srow][32 * sq];
      float ps = 0.0f;
#pragma unroll
      for (int g = 0; g < 4; ++g) {
        v4f a = scv[2 * g];
        v4f c = scv[2 * g + 1];
        v8h pk;
#pragma unroll
        for (int i = 0; i < 4; ++i) {
          const float p0 = __builtin_amdgcn_exp2f(a[i] - nm);
          const float p1 = __builtin_amdgcn_exp2f(c[i] - nm);
          pk[i]     = (_Float16)p0;
          pk[i + 4] = (_Float16)p1;
          ps += p0 + p1;
        }
        // reorder: first 4 from a (cols 8g..), next 4 from c (cols 8g+4..)
        prow[g] = pk;
      }
      psum[srow][sq] = ps;
    }
    __syncthreads();

    // --- softmax, step 3: running-stat update (one thread per row) ---
    if (tid < 64) {
      const int row = tid;
      const float nm = fmaxf(rm_s[row],
                             fmaxf(fmaxf(pmax[row][0], pmax[row][1]),
                                   fmaxf(pmax[row][2], pmax[row][3])));
      const float corr = __builtin_amdgcn_exp2f(rm_s[row] - nm);
      rs_s[row] = rs_s[row] * corr +
                  (psum[row][0] + psum[row][1] + psum[row][2] + psum[row][3]);
      rm_s[row] = nm;
      corr_s[row] = corr;
    }
    __syncthreads();

    // --- rescale running PV accumulator, then PV: out += P * V ---
#pragma unroll
    for (int j = 0; j < 8; ++j) oacc[j] *= corr_s[16 * r + j + hi8];
#pragma unroll
    for (int ks = 0; ks < 4; ++ks) {
      v16h pf = frag_a_f16(&pbuf[16 * r][32 * ks], kPPAD);
      v16h vf = frag_b_f16(Vb + (size_t)(16 * cq) * kS + (t0 + 32 * ks), kS);
      oacc = wmma16(pf, vf, oacc);
    }
    __syncthreads();  // protect pbuf/corr_s/sc/rm/rs before next tile
  }

  // One reciprocal per row, then normalize + store to AO [B,S,C] f16.
  if (tid < 64) corr_s[tid] = 1.0f / rs_s[tid];
  __syncthreads();
  const int b = bh / kNH, nh = bh % kNH;
#pragma unroll
  for (int j = 0; j < 8; ++j) {
    const int row = 16 * r + j + hi8;
    const float v = oacc[j] * corr_s[row];
    const int s = s0 + row;
    AO[((size_t)b * kS + s) * kC + nh * kHD + 16 * cq + ncol] = (_Float16)v;
  }
}

// NOTE on pk ordering in step 2: pk[0..3] hold cols 8g+0..3, pk[4..7] hold
// cols 8g+4..7, matching prow[g] = pbuf[srow][32*sq + 8g .. +8g+7]. The two
// v4f loads cover exactly those 8 consecutive columns.

// ---------------------------------------------------------------------------
// Kernel 3: output projection + bias + residual, f32 out [B,C,S].
// ---------------------------------------------------------------------------
__global__ __launch_bounds__(256) void oproj_kernel(
    const _Float16* __restrict__ AO, const float* __restrict__ Wo,
    const float* __restrict__ bo, const float* __restrict__ x,
    float* __restrict__ out) {
  const int lane  = threadIdx.x & 31;
  const int w     = threadIdx.x >> 5;
  const int r     = w & 3;
  const int cpair = w >> 2;
  const int s0 = blockIdx.x * 64;
  const int o0 = blockIdx.y * 64 + 16 * r;
  const int b  = blockIdx.z;
  const _Float16* ab = AO + (size_t)b * kS * kC;

  v8f acc0 = {}, acc1 = {};
  for (int kk = 0; kk < kC; kk += 32) {
    v16h fa  = frag_a_f32(Wo + o0 * kC + kk, kC);
    v16h fb0 = frag_b_f16(ab + (s0 + 32 * cpair) * kC + kk, kC);
    v16h fb1 = frag_b_f16(ab + (s0 + 32 * cpair + 16) * kC + kk, kC);
    acc0 = wmma16(fa, fb0, acc0);
    acc1 = wmma16(fa, fb1, acc1);
  }

  const int hi8  = (lane >> 4) << 3;
  const int ncol = lane & 15;
#pragma unroll
  for (int j = 0; j < 8; ++j) {
    const int o = o0 + j + hi8;
    const float bias = bo[o];
#pragma unroll
    for (int ct = 0; ct < 2; ++ct) {
      const int s = s0 + 32 * cpair + 16 * ct + ncol;
      const size_t idx = ((size_t)b * kC + o) * kS + s;
      out[idx] = (ct ? acc1[j] : acc0[j]) + bias + x[idx];
    }
  }
}

// ---------------------------------------------------------------------------
extern "C" void kernel_launch(void* const* d_in, const int* in_sizes, int n_in,
                              void* d_out, int out_size, void* d_ws, size_t ws_size,
                              hipStream_t stream) {
  const float* x  = (const float*)d_in[0];
  const float* Wq = (const float*)d_in[1];
  const float* bq = (const float*)d_in[2];
  const float* Wk = (const float*)d_in[3];
  const float* bk = (const float*)d_in[4];
  const float* Wv = (const float*)d_in[5];
  const float* bv = (const float*)d_in[6];
  const float* Wo = (const float*)d_in[7];
  const float* bo = (const float*)d_in[8];
  float* out = (float*)d_out;

  const size_t N = (size_t)kB * kS * kC;  // elements per f16 tensor
  _Float16* xt = (_Float16*)d_ws;         // [B,S,C]
  _Float16* Qd = xt + N;                  // [B,NH,S,HD]
  _Float16* Kd = Qd + N;                  // [B,NH,S,HD]
  _Float16* Vt = Kd + N;                  // [B,NH,HD,S]
  _Float16* AO = xt;                      // reuse xt after qkv: [B,S,C]

  xpose_kernel<<<dim3(kS / 32, kC / 32, kB), dim3(32, 8), 0, stream>>>(x, xt);
  qkv_kernel<<<dim3(kS / 64, kC / 64, kB * 3), 256, 0, stream>>>(
      xt, Wq, bq, Wk, bk, Wv, bv, Qd, Kd, Vt);
  attn_kernel<<<dim3(kS / 64, kB * kNH), 256, 0, stream>>>(Qd, Kd, Vt, AO);
  oproj_kernel<<<dim3(kS / 64, kC / 64, kB), 256, 0, stream>>>(AO, Wo, bo, x, out);
}
